// OnsetsAndFrames_with_fast_local_attn_41566693491392
// MI455X (gfx1250) — matrix operand order repaired and checked by reading, not compiled, not run on hardware
//
#include <hip/hip_runtime.h>
#include <hip/hip_bf16.h>
#include <math.h>

typedef unsigned short u16;
typedef __attribute__((ext_vector_type(16))) __bf16 v16bf;
typedef __attribute__((ext_vector_type(8)))  float  v8f;

__device__ __forceinline__ u16 f2bf(float f) {
  unsigned x = __float_as_uint(f);
  unsigned r = x + 0x7FFFu + ((x >> 16) & 1u);
  return (u16)(r >> 16);
}
__device__ __forceinline__ float bf2f(u16 h) {
  return __uint_as_float(((unsigned)h) << 16);
}

union Frag { v16bf v; uint4 q[2]; };

// ---------------------------------------------------------------------------
// BN folding: scale = g/sqrt(v+eps); shift = conv_b*scale + b - m*scale
// ---------------------------------------------------------------------------
__global__ void bn_prep(const float* __restrict__ g, const float* __restrict__ bb,
                        const float* __restrict__ mm, const float* __restrict__ vv,
                        const float* __restrict__ cb, int C,
                        float* __restrict__ scale, float* __restrict__ shift) {
  int c = blockIdx.x * blockDim.x + threadIdx.x;
  if (c >= C) return;
  float s = g[c] * rsqrtf(vv[c] + 1e-5f);
  scale[c] = s;
  shift[c] = cb[c] * s + bb[c] - mm[c] * s;
}

// ---------------------------------------------------------------------------
// conv1: 1->48ch 3x3 SAME + BN + ReLU, direct VALU (1 GFLOP).
// Output channels-last padded: x1[b][t][f][64] bf16 (c>=48 -> 0).
// ---------------------------------------------------------------------------
__global__ void conv1_kernel(const float* __restrict__ spec, const float* __restrict__ w,
                             const float* __restrict__ scale, const float* __restrict__ shift,
                             u16* __restrict__ x1) {
  const int T = 640, F = 229;
  int idx = blockIdx.x * blockDim.x + threadIdx.x;
  int total = 8 * T * F * 64;
  if (idx >= total) return;
  int c = idx & 63; int r = idx >> 6;
  int f = r % F; r /= F;
  int t = r % T; int b = r / T;
  u16 res = 0;
  if (c < 48) {
    float acc = 0.f;
    const float* wc = w + c * 9;
    #pragma unroll
    for (int dt = 0; dt < 3; ++dt) {
      int tt = t + dt - 1;
      if (tt < 0 || tt >= T) continue;
      #pragma unroll
      for (int df = 0; df < 3; ++df) {
        int ff = f + df - 1;
        if (ff < 0 || ff >= F) continue;
        acc += spec[(b * T + tt) * F + ff] * wc[dt * 3 + df];
      }
    }
    res = f2bf(fmaxf(acc * scale[c] + shift[c], 0.f));
  }
  x1[idx] = res;
}

// ---------------------------------------------------------------------------
// Pack linear weight W[N][ldw] (cols colOff..colOff+Ka) into WMMA B-fragment
// order: Bp[((kb*NT + nt)*32 + lane)*16 + j], lane holds col nt*16+(lane&15),
// rows k = kb*32 + (lane>>4)*16 + j.  Zero-filled outside N/Ka.
// ---------------------------------------------------------------------------
__global__ void pack_frag_lin(const float* __restrict__ w, int N, int Ka, int ldw, int colOff,
                              int NT, int steps, u16* __restrict__ Bp) {
  int idx = blockIdx.x * blockDim.x + threadIdx.x;
  int total = steps * NT * 512;
  if (idx >= total) return;
  int j = idx & 15; int lane = (idx >> 4) & 31; int rest = idx >> 9;
  int nt = rest % NT; int kb = rest / NT;
  int n = nt * 16 + (lane & 15);
  int k = kb * 32 + (lane >> 4) * 16 + j;
  u16 v = 0;
  if (n < N && k < Ka) v = f2bf(w[n * ldw + colOff + k]);
  Bp[idx] = v;
}

// Conv weight w[O][I][3][3] -> fragment order with K = tap*64 + c (c padded to 64).
__global__ void pack_frag_conv(const float* __restrict__ w, int O, int I,
                               int NT, u16* __restrict__ Bp) {
  int idx = blockIdx.x * blockDim.x + threadIdx.x;
  int total = 18 * NT * 512;       // steps = 576/32 = 18
  if (idx >= total) return;
  int j = idx & 15; int lane = (idx >> 4) & 31; int rest = idx >> 9;
  int nt = rest % NT; int kb = rest / NT;
  int n = nt * 16 + (lane & 15);
  int k = kb * 32 + (lane >> 4) * 16 + j;     // 0..575
  int tap = k >> 6, c = k & 63;
  u16 v = 0;
  if (n < O && c < I) v = f2bf(w[(n * I + c) * 9 + tap]);
  Bp[idx] = v;
}

// ---------------------------------------------------------------------------
// Plain WMMA GEMM, no LDS: A[M][Ka32] bf16 row-major (Ka32 % 32 == 0, zero
// padded), Bp in fragment order. M % 64 == 0. Block = 8 waves, tile 64x64,
// wave computes 16x32 (2 wmma / K-step). Fragments loaded with b128s.
// epi: 0=raw f32, 2=bias+sigmoid, 3=bias
// ---------------------------------------------------------------------------
__launch_bounds__(256)
__global__ void gemm_plain_wmma(const u16* __restrict__ A, const u16* __restrict__ Bp,
                                int M, int N, int Ka32, int NT,
                                const float* __restrict__ bias, int epi,
                                float* __restrict__ outF, u16* __restrict__ outB,
                                int ldF, int ldB, int NpadB) {
  int tid = threadIdx.x, lane = tid & 31, wave = tid >> 5;
  int wm = wave >> 1, wn = wave & 1;
  int half = lane >> 4, l16 = lane & 15;
  int mBase = blockIdx.x * 64, nBase = blockIdx.y * 64;
  int m = mBase + wm * 16 + l16;
  int nt0 = (nBase >> 4) + wn * 2;
  int nt1 = nt0 + 1;
  if (nt0 >= NT) nt0 = 0;
  if (nt1 >= NT) nt1 = 0;
  v8f c0 = {}, c1 = {};
  const u16* arow = A + (size_t)m * Ka32 + half * 16;
  const u16* bl = Bp + (size_t)lane * 16;
  int steps = Ka32 >> 5;
  #pragma unroll 2
  for (int kb = 0; kb < steps; ++kb) {
    Frag af, b0, b1;
    const uint4* ap = (const uint4*)(arow + kb * 32);
    af.q[0] = ap[0];
    af.q[1] = ap[1];
    const uint4* bp0 = (const uint4*)(bl + ((size_t)kb * NT + nt0) * 512);
    b0.q[0] = bp0[0]; b0.q[1] = bp0[1];
    const uint4* bp1 = (const uint4*)(bl + ((size_t)kb * NT + nt1) * 512);
    b1.q[0] = bp1[0]; b1.q[1] = bp1[1];
    c0 = __builtin_amdgcn_wmma_f32_16x16x32_bf16(false, af.v, false, b0.v, (short)0, c0, false, false);
    c1 = __builtin_amdgcn_wmma_f32_16x16x32_bf16(false, af.v, false, b1.v, (short)0, c1, false, false);
  }
  int mRow = mBase + wm * 16 + half * 8;
  #pragma unroll
  for (int r = 0; r < 8; ++r) {
    int mo = mRow + r;
    #pragma unroll
    for (int s = 0; s < 2; ++s) {
      int n = nBase + wn * 32 + s * 16 + l16;
      float acc = s ? c1[r] : c0[r];
      float o;
      if (epi == 2)      o = 1.f / (1.f + __expf(-(acc + bias[n < N ? n : 0])));
      else if (epi == 3) o = acc + bias[n < N ? n : 0];
      else               o = acc;
      if (outF && n < N) outF[(size_t)mo * ldF + n] = o;
      if (outB && n < NpadB) outB[(size_t)mo * ldB + n] = (n < N) ? f2bf(o) : (u16)0;
    }
  }
}

// ---------------------------------------------------------------------------
// Conv WMMA GEMM (3x3 SAME as 9 accumulated GEMMs), no LDS.
// X[b][T][F][64] bf16 channels-last padded; rows m=(b,t,f), M % 64 == 0.
// K = 9 taps * 64ch = 576 (18 K-steps). Epilogue: BN + ReLU -> bf16.
// ---------------------------------------------------------------------------
__launch_bounds__(256)
__global__ void gemm_conv_wmma(const u16* __restrict__ X, const u16* __restrict__ Bp,
                               int M, int N, int T, int F, int NT,
                               const float* __restrict__ scale, const float* __restrict__ shift,
                               u16* __restrict__ outB, int ldB) {
  int tid = threadIdx.x, lane = tid & 31, wave = tid >> 5;
  int wm = wave >> 1, wn = wave & 1;
  int half = lane >> 4, l16 = lane & 15;
  int mBase = blockIdx.x * 64, nBase = blockIdx.y * 64;
  int m = mBase + wm * 16 + l16;
  int f = m % F; int rem = m / F;
  int t = rem % T; int b = rem / T;
  int nt0 = (nBase >> 4) + wn * 2;
  int nt1 = nt0 + 1;
  if (nt0 >= NT) nt0 = 0;
  if (nt1 >= NT) nt1 = 0;
  v8f c0 = {}, c1 = {};
  const u16* bl = Bp + (size_t)lane * 16;
  #pragma unroll
  for (int tap = 0; tap < 9; ++tap) {
    int dt = tap / 3, df = tap % 3;
    int tt = t + dt - 1, ff = f + df - 1;
    bool valid = (tt >= 0 && tt < T && ff >= 0 && ff < F);
    size_t rowoff = 0;
    if (valid) rowoff = (((size_t)(b * T + tt) * F + ff) << 6);
    #pragma unroll
    for (int kc = 0; kc < 2; ++kc) {
      Frag af, b0, b1;
      uint4 z; z.x = z.y = z.z = z.w = 0u;
      af.q[0] = z; af.q[1] = z;
      if (valid) {
        const uint4* ap = (const uint4*)(X + rowoff + kc * 32 + half * 16);
        af.q[0] = ap[0];
        af.q[1] = ap[1];
      }
      int kb = tap * 2 + kc;
      const uint4* bp0 = (const uint4*)(bl + ((size_t)kb * NT + nt0) * 512);
      b0.q[0] = bp0[0]; b0.q[1] = bp0[1];
      const uint4* bp1 = (const uint4*)(bl + ((size_t)kb * NT + nt1) * 512);
      b1.q[0] = bp1[0]; b1.q[1] = bp1[1];
      c0 = __builtin_amdgcn_wmma_f32_16x16x32_bf16(false, af.v, false, b0.v, (short)0, c0, false, false);
      c1 = __builtin_amdgcn_wmma_f32_16x16x32_bf16(false, af.v, false, b1.v, (short)0, c1, false, false);
    }
  }
  int mRow = mBase + wm * 16 + half * 8;
  #pragma unroll
  for (int r = 0; r < 8; ++r) {
    int mo = mRow + r;
    #pragma unroll
    for (int s = 0; s < 2; ++s) {
      int n = nBase + wn * 32 + s * 16 + l16;
      if (n >= N) continue;
      float acc = s ? c1[r] : c0[r];
      float o = fmaxf(acc * scale[n] + shift[n], 0.f);
      outB[(size_t)mo * ldB + n] = f2bf(o);
    }
  }
}

// ---------------------------------------------------------------------------
// Freq max-pools (ReLU already applied; max commutes)
// ---------------------------------------------------------------------------
__global__ void pool1_kernel(const u16* __restrict__ y2, u16* __restrict__ x2) {
  // y2: [(b,t,f229)][48] -> x2: [b][t][f114][64] (padded)
  int idx = blockIdx.x * blockDim.x + threadIdx.x;
  int total = 8 * 640 * 114 * 64;
  if (idx >= total) return;
  int c = idx & 63; int r = idx >> 6;
  int fp = r % 114; r /= 114;
  int t = r % 640; int b = r / 640;
  u16 res = 0;
  if (c < 48) {
    size_t base = ((size_t)(b * 640 + t) * 229 + 2 * fp) * 48 + c;
    res = f2bf(fmaxf(bf2f(y2[base]), bf2f(y2[base + 48])));
  }
  x2[idx] = res;
}

__global__ void pool2_kernel(const u16* __restrict__ y3, u16* __restrict__ feats) {
  // y3: [(b,t,f114)][96] -> feats: [(b,t)][c*57+f]  (5472 dense, K%32==0)
  int idx = blockIdx.x * blockDim.x + threadIdx.x;
  int total = 8 * 640 * 96 * 57;
  if (idx >= total) return;
  int fp = idx % 57; int r = idx / 57;
  int c = r % 96; r /= 96;
  int t = r % 640; int b = r / 640;
  size_t base = ((size_t)(b * 640 + t) * 114 + 2 * fp) * 96 + c;
  float a0 = bf2f(y3[base]);
  float a1 = bf2f(y3[base + 96]);
  feats[(size_t)(b * 640 + t) * 5472 + c * 57 + fp] = f2bf(fmaxf(a0, a1));
}

// ---------------------------------------------------------------------------
// combined (ld 192, zero-padded) = [onset(88) | activation(88) | 0...]
// cat2 (ld 288, zero-padded)     = [combined(176) | weighted(88) | 0...]
// predB buffers have ld 96 (cols 88..95 zeroed by GEMM epilogue).
// ---------------------------------------------------------------------------
__global__ void build_combined(const u16* __restrict__ on, const u16* __restrict__ ac,
                               u16* __restrict__ comb) {
  int idx = blockIdx.x * blockDim.x + threadIdx.x;
  if (idx >= 5120 * 192) return;
  int n = idx % 192; int m = idx / 192;
  u16 v = 0;
  if (n < 88)       v = on[m * 96 + n];
  else if (n < 176) v = ac[m * 96 + (n - 88)];
  comb[idx] = v;
}

__global__ void build_cat2(const u16* __restrict__ comb, const float* __restrict__ wgt,
                           u16* __restrict__ cat2) {
  int idx = blockIdx.x * blockDim.x + threadIdx.x;
  if (idx >= 5120 * 288) return;
  int n = idx % 288; int m = idx / 288;
  u16 v = 0;
  if (n < 176)      v = comb[m * 192 + n];
  else if (n < 264) v = f2bf(wgt[m * 88 + (n - 176)]);
  cat2[idx] = v;
}

// ---------------------------------------------------------------------------
// Local attention (factorized): energy[b,t,w] = tanh(hidW[b,t]+encW[b,t+w-30]+b)
// One 128-thread block per (b,t).
// ---------------------------------------------------------------------------
__global__ void attn_kernel(const float* __restrict__ hidW, const float* __restrict__ encW,
                            const float* __restrict__ act, const float* __restrict__ attn_b,
                            const float* __restrict__ v_w,
                            float* __restrict__ a_out, float* __restrict__ weighted) {
  __shared__ float sc[61];
  __shared__ float a_sh[61];
  int m = blockIdx.x;
  int b = m / 640, t = m % 640;
  int tid = threadIdx.x;
  if (tid < 61) {
    int tp = t + tid - 30;
    bool in = (tp >= 0 && tp < 640);
    const float* e = encW + (size_t)(b * 640 + (in ? tp : t)) * 88;
    float msk = in ? 1.f : 0.f;
    const float* h = hidW + (size_t)m * 88;
    float s = 0.f;
    for (int n = 0; n < 88; ++n)
      s += v_w[n] * tanhf(h[n] + attn_b[n] + msk * e[n]);
    sc[tid] = s;
  }
  __syncthreads();
  if (tid == 0) {
    float mx = -1e30f;
    for (int w = 0; w < 61; ++w) mx = fmaxf(mx, sc[w]);
    float sum = 0.f;
    for (int w = 0; w < 61; ++w) { float e = __expf(sc[w] - mx); a_sh[w] = e; sum += e; }
    float inv = 1.f / sum;
    for (int w = 0; w < 61; ++w) a_sh[w] *= inv;
  }
  __syncthreads();
  if (tid < 61) a_out[(size_t)m * 61 + tid] = a_sh[tid];
  if (tid < 88) {
    float acc = 0.f;
    for (int w = 0; w < 61; ++w) {
      int tp = t + w - 30;
      if (tp >= 0 && tp < 640) acc += a_sh[w] * act[(size_t)(b * 640 + tp) * 88 + tid];
    }
    weighted[(size_t)m * 88 + tid] = acc;
  }
}

// ---------------------------------------------------------------------------
static inline int ceilDiv(int a, int b) { return (a + b - 1) / b; }

extern "C" void kernel_launch(void* const* d_in, const int* in_sizes, int n_in,
                              void* d_out, int out_size, void* d_ws, size_t ws_size,
                              hipStream_t stream) {
  (void)in_sizes; (void)n_in; (void)out_size; (void)ws_size;
  const float* spec = (const float*)d_in[0];
  auto P = [&](int i) { return (const float*)d_in[i]; };

  char* ws = (char*)d_ws;
  size_t off = 0;
  auto AL = [&](size_t bytes) { size_t r = off; off += (bytes + 255) & ~(size_t)255; return r; };

  const int M2 = 8 * 640 * 229;   // 1,172,480 (= 64*18320)
  const int M3 = 8 * 640 * 114;   //   583,680 (= 64*9120)
  const int MT = 8 * 640;         //     5,120 (= 64*80)

  size_t oR0   = AL((size_t)8 * 640 * 229 * 64 * 2);  // x1 (150MB) / y3 (112MB)
  size_t oR1   = AL((size_t)M2 * 48 * 2);             // y2 (112MB) / feats (56MB)
  size_t oX2   = AL((size_t)8 * 640 * 114 * 64 * 2);  // 75MB
  size_t oHf   = AL((size_t)MT * 768 * 2);
  size_t oBtFC = AL((size_t)171 * 48 * 512 * 2);
  size_t oBtC2 = AL((size_t)18 * 4 * 512 * 2);
  size_t oBtC3 = AL((size_t)18 * 8 * 512 * 2);
  size_t oBtH  = AL((size_t)24 * 8 * 512 * 2);
  size_t oPr0  = AL((size_t)MT * 96 * 2);
  size_t oPr1  = AL((size_t)MT * 96 * 2);
  size_t oComb = AL((size_t)MT * 192 * 2);
  size_t oCat2 = AL((size_t)MT * 288 * 2);
  size_t oHidW = AL((size_t)MT * 88 * 4);
  size_t oEncW = AL((size_t)MT * 88 * 4);
  size_t oWgt  = AL((size_t)MT * 88 * 4);
  size_t oBtA  = AL((size_t)6 * 8 * 512 * 2);
  size_t oBtE  = AL((size_t)3 * 8 * 512 * 2);
  size_t oBtCo = AL((size_t)9 * 8 * 512 * 2);
  size_t oSc1 = AL(96 * 4), oSh1 = AL(96 * 4);
  size_t oSc2 = AL(96 * 4), oSh2 = AL(96 * 4);
  size_t oSc3 = AL(96 * 4), oSh3 = AL(96 * 4);

  u16* x1    = (u16*)(ws + oR0);
  u16* y3    = (u16*)(ws + oR0);   // alias: x1 dead before conv3 GEMM writes y3
  u16* y2    = (u16*)(ws + oR1);
  u16* feats = (u16*)(ws + oR1);   // alias: y2 dead before pool2 writes feats
  u16* x2    = (u16*)(ws + oX2);
  u16* hfeat = (u16*)(ws + oHf);
  u16* BtFC  = (u16*)(ws + oBtFC);
  u16* BtC2  = (u16*)(ws + oBtC2);
  u16* BtC3  = (u16*)(ws + oBtC3);
  u16* BtH   = (u16*)(ws + oBtH);
  u16* predB[2] = { (u16*)(ws + oPr0), (u16*)(ws + oPr1) };
  u16* comb  = (u16*)(ws + oComb);
  u16* cat2  = (u16*)(ws + oCat2);
  float* hidW = (float*)(ws + oHidW);
  float* encW = (float*)(ws + oEncW);
  float* wgt  = (float*)(ws + oWgt);
  u16* BtA  = (u16*)(ws + oBtA);
  u16* BtE  = (u16*)(ws + oBtE);
  u16* BtCo = (u16*)(ws + oBtCo);
  float* sc1 = (float*)(ws + oSc1); float* sh1 = (float*)(ws + oSh1);
  float* sc2 = (float*)(ws + oSc2); float* sh2 = (float*)(ws + oSh2);
  float* sc3 = (float*)(ws + oSc3); float* sh3 = (float*)(ws + oSh3);

  float* out = (float*)d_out;
  float* outOnset = out;            // 5120*88
  float* outAct   = out + 450560;   // 5120*88
  float* outFrame = out + 901120;   // 5120*88
  float* outA     = out + 1351680;  // 5120*61

  // ------------- two conv stacks (onset: s=0, frame: s=1) -------------
  for (int s = 0; s < 2; ++s) {
    int base = 1 + s * 20;
    const float *c1w = P(base+0),  *c1b = P(base+1);
    const float *bn1g = P(base+2), *bn1b = P(base+3), *bn1m = P(base+4), *bn1v = P(base+5);
    const float *c2w = P(base+6),  *c2b = P(base+7);
    const float *bn2g = P(base+8), *bn2b = P(base+9), *bn2m = P(base+10), *bn2v = P(base+11);
    const float *c3w = P(base+12), *c3b = P(base+13);
    const float *bn3g = P(base+14),*bn3b = P(base+15),*bn3m = P(base+16), *bn3v = P(base+17);
    const float *fcw = P(base+18), *fcb = P(base+19);

    bn_prep<<<1, 64, 0, stream>>>(bn1g, bn1b, bn1m, bn1v, c1b, 48, sc1, sh1);
    bn_prep<<<1, 64, 0, stream>>>(bn2g, bn2b, bn2m, bn2v, c2b, 48, sc2, sh2);
    bn_prep<<<1, 128, 0, stream>>>(bn3g, bn3b, bn3m, bn3v, c3b, 96, sc3, sh3);

    conv1_kernel<<<ceilDiv(8*640*229*64, 256), 256, 0, stream>>>(spec, c1w, sc1, sh1, x1);

    pack_frag_conv<<<ceilDiv(18*4*512, 256), 256, 0, stream>>>(c2w, 48, 48, 4, BtC2);
    gemm_conv_wmma<<<dim3(M2/64, 1), 256, 0, stream>>>(x1, BtC2, M2, 48, 640, 229, 4,
                                                       sc2, sh2, y2, 48);

    pool1_kernel<<<ceilDiv(8*640*114*64, 256), 256, 0, stream>>>(y2, x2);

    pack_frag_conv<<<ceilDiv(18*8*512, 256), 256, 0, stream>>>(c3w, 96, 48, 8, BtC3);
    gemm_conv_wmma<<<dim3(M3/64, 2), 256, 0, stream>>>(x2, BtC3, M3, 96, 640, 114, 8,
                                                       sc3, sh3, y3, 96);

    pool2_kernel<<<ceilDiv(8*640*96*57, 256), 256, 0, stream>>>(y3, feats);

    pack_frag_lin<<<ceilDiv(171*48*512, 256), 256, 0, stream>>>(fcw, 768, 5472, 5472, 0,
                                                                48, 171, BtFC);
    gemm_plain_wmma<<<dim3(MT/64, 12), 256, 0, stream>>>(feats, BtFC, MT, 768, 5472, 48,
                                                         fcb, /*bias*/3,
                                                         nullptr, hfeat, 0, 768, 768);

    const float* hw = (s == 0) ? P(41) : P(43);
    const float* hb = (s == 0) ? P(42) : P(44);
    pack_frag_lin<<<ceilDiv(24*8*512, 256), 256, 0, stream>>>(hw, 88, 768, 768, 0, 8, 24, BtH);
    gemm_plain_wmma<<<dim3(MT/64, 2), 256, 0, stream>>>(hfeat, BtH, MT, 88, 768, 8,
                                                        hb, /*bias+sigmoid*/2,
                                                        (s == 0) ? outOnset : outAct,
                                                        predB[s], 88, 96, 96);
  }

  // ------------- local attention -------------
  const float* attn_w = P(45); const float* attn_b = P(46);
  const float* v_w    = P(47);
  const float* comb_w = P(48); const float* comb_b = P(49);

  build_combined<<<ceilDiv(5120*192, 256), 256, 0, stream>>>(predB[0], predB[1], comb);

  pack_frag_lin<<<ceilDiv(6*8*512, 256), 256, 0, stream>>>(attn_w, 88, 176, 264, 0, 8, 6, BtA);
  gemm_plain_wmma<<<dim3(MT/64, 2), 256, 0, stream>>>(comb, BtA, MT, 88, 192, 8,
                                                      nullptr, /*raw*/0,
                                                      hidW, nullptr, 88, 0, 0);

  pack_frag_lin<<<ceilDiv(3*8*512, 256), 256, 0, stream>>>(attn_w, 88, 88, 264, 176, 8, 3, BtE);
  gemm_plain_wmma<<<dim3(MT/64, 2), 256, 0, stream>>>(predB[1], BtE, MT, 88, 96, 8,
                                                      nullptr, /*raw*/0,
                                                      encW, nullptr, 88, 0, 0);

  attn_kernel<<<MT, 128, 0, stream>>>(hidW, encW, outAct, attn_b, v_w, outA, wgt);

  build_cat2<<<ceilDiv(5120*288, 256), 256, 0, stream>>>(comb, wgt, cat2);

  pack_frag_lin<<<ceilDiv(9*8*512, 256), 256, 0, stream>>>(comb_w, 88, 264, 264, 0, 8, 9, BtCo);
  gemm_plain_wmma<<<dim3(MT/64, 2), 256, 0, stream>>>(cat2, BtCo, MT, 88, 288, 8,
                                                      comb_b, /*bias+sigmoid*/2,
                                                      outFrame, nullptr, 88, 0, 0);
}
